// ChaosKernel_30021821399810
// MI455X (gfx1250) — compile-verified, matrix operand order checked by reading, not run on hardware
//
#include <hip/hip_runtime.h>
#include <hip/hip_bf16.h>

// Problem constants (from reference)
#define B_ 2
#define T_ 512
#define D_ 64
#define H_ 32
#define L_ 4
#define NPASS_ 2
#define EPS_ 1e-8f

typedef _Float16 half_t;
typedef __attribute__((ext_vector_type(16))) _Float16 v16h;
typedef __attribute__((ext_vector_type(8)))  _Float16 v8h;
typedef __attribute__((ext_vector_type(8)))  float    v8f;

__device__ __forceinline__ float sigmoidf_(float z) {
    return 1.0f / (1.0f + __expf(-z));
}

// ---------------------------------------------------------------------------
// temps[l] = sigmoid(dot(W_temp[l], basin) + b_temp[l]) + 0.5   (one block, 64 thr)
// ---------------------------------------------------------------------------
__global__ void temps_kernel(const float* __restrict__ W_temp,
                             const float* __restrict__ b_temp,
                             const float* __restrict__ basin,
                             float* __restrict__ temps) {
    int d = threadIdx.x;              // 0..63
    __shared__ float red[D_];
    float bv = basin[d];
    for (int l = 0; l < L_; ++l) {
        red[d] = W_temp[l * D_ + d] * bv;
        __syncthreads();
        for (int s = 32; s > 0; s >>= 1) {
            if (d < s) red[d] += red[d + s];
            __syncthreads();
        }
        if (d == 0) temps[l] = sigmoidf_(red[0] + b_temp[l]) + 0.5f;
        __syncthreads();
    }
}

// ---------------------------------------------------------------------------
// Feedback gate: x = x*g + prev*(1-g), g = sigmoid([x,prev] @ W_fb[l]^T + b_fb[l])
// one block per (b,t) row, 64 threads (one per output dim)
// ---------------------------------------------------------------------------
__global__ void feedback_gate(float* x, const float* __restrict__ prev,
                              const float* __restrict__ W_fb,
                              const float* __restrict__ b_fb, int l) {
    int row = blockIdx.x;             // b*T + t
    int d = threadIdx.x;              // 0..63
    __shared__ float xs[D_], ps[D_];
    size_t base = (size_t)row * D_;
    xs[d] = x[base + d];
    ps[d] = prev[base + d];
    __syncthreads();
    const float* Wl = W_fb + ((size_t)l * D_ + d) * (2 * D_);
    float acc = b_fb[l * D_ + d];
    #pragma unroll 8
    for (int j = 0; j < D_; ++j) acc += xs[j] * Wl[j] + ps[j] * Wl[D_ + j];
    float g = sigmoidf_(acc);
    x[base + d] = xs[d] * g + ps[d] * (1.0f - g);
}

// ---------------------------------------------------------------------------
// Simplex projection: p = softplus(x) normalized twice, sp = sqrt(p) (f16);
// also stash xh = (f16)x for the second GEMM. One block per row, 64 threads.
// ---------------------------------------------------------------------------
__global__ void simplex_prep(const float* __restrict__ x,
                             half_t* __restrict__ sp,
                             half_t* __restrict__ xh) {
    int row = blockIdx.x;
    int d = threadIdx.x;
    __shared__ float red[D_];
    size_t idx = (size_t)row * D_ + d;
    float v = x[idx];
    float s = (v > 20.0f) ? v : log1pf(__expf(v));   // softplus
    red[d] = s;
    __syncthreads();
    for (int st = 32; st > 0; st >>= 1) {
        if (d < st) red[d] += red[d + st];
        __syncthreads();
    }
    float sum1 = red[0];
    __syncthreads();
    float p = fmaxf(s / (sum1 + EPS_), EPS_);
    red[d] = p;
    __syncthreads();
    for (int st = 32; st > 0; st >>= 1) {
        if (d < st) red[d] += red[d + st];
        __syncthreads();
    }
    float sum2 = red[0];
    p = p / (sum2 + EPS_);
    sp[idx] = (half_t)sqrtf(p);
    xh[idx] = (half_t)v;
}

// ---------------------------------------------------------------------------
// GEMM 1 (WMMA): S = SP · SP^T, fused epilogue -> logits = -2*acos(clip(S))/temp
// 4 waves/block, each wave owns one 16x16 tile. K = 64 (2 WMMA steps).
// Both operands are contiguous per-lane rows of SP -> 16B vector loads.
// ---------------------------------------------------------------------------
__global__ void qfi_logits_gemm(const half_t* __restrict__ sp,
                                const float* __restrict__ temps, int l,
                                float* __restrict__ logits) {
    int wid  = threadIdx.x >> 5;
    int lane = threadIdx.x & 31;
    int tile = blockIdx.x * 4 + wid;              // 0 .. B*32*32-1
    int jt = tile & 31; tile >>= 5;
    int it = tile & 31; int b = tile >> 5;
    int m  = lane & 15, hi = lane >> 4;

    const half_t* spb  = sp + (size_t)b * T_ * D_;
    const half_t* arow = spb + (size_t)(it * 16 + m) * D_;
    const half_t* brow = spb + (size_t)(jt * 16 + m) * D_;   // n == lane&15 == m

    v8f c = {};
    #pragma unroll
    for (int k0 = 0; k0 < D_; k0 += 32) {
        v8h alo = *(const v8h*)(arow + k0 + hi * 8);
        v8h ahi = *(const v8h*)(arow + k0 + 16 + hi * 8);
        v8h blo = *(const v8h*)(brow + k0 + hi * 8);
        v8h bhi = *(const v8h*)(brow + k0 + 16 + hi * 8);
        v16h a, bb;
        #pragma unroll
        for (int e = 0; e < 8; ++e) {
            a[e] = alo[e];  a[e + 8] = ahi[e];
            bb[e] = blo[e]; bb[e + 8] = bhi[e];
        }
        c = __builtin_amdgcn_wmma_f32_16x16x32_f16(false, a, false, bb,
                                                   (short)0, c, false, false);
    }
    float invT = 1.0f / fmaxf(temps[l], 1e-6f);
    #pragma unroll
    for (int r = 0; r < 8; ++r) {
        int M = r + hi * 8;
        float inner = fminf(fmaxf(c[r], -1.0f + 1e-6f), 1.0f - 1e-6f);
        logits[((size_t)b * T_ + it * 16 + M) * T_ + jt * 16 + m] =
            -2.0f * acosf(inner) * invT;
    }
}

// ---------------------------------------------------------------------------
// Row softmax over T=512 -> f16 attention weights. 256 threads/row.
// ---------------------------------------------------------------------------
__global__ void softmax_rows(const float* __restrict__ logits,
                             half_t* __restrict__ w) {
    int row = blockIdx.x;
    int tid = threadIdx.x;            // 0..255
    const float* Lr = logits + (size_t)row * T_;
    float v0 = Lr[tid], v1 = Lr[tid + 256];
    __shared__ float red[256];
    red[tid] = fmaxf(v0, v1);
    __syncthreads();
    for (int s = 128; s > 0; s >>= 1) {
        if (tid < s) red[tid] = fmaxf(red[tid], red[tid + s]);
        __syncthreads();
    }
    float mx = red[0];
    __syncthreads();
    float e0 = __expf(v0 - mx), e1 = __expf(v1 - mx);
    red[tid] = e0 + e1;
    __syncthreads();
    for (int s = 128; s > 0; s >>= 1) {
        if (tid < s) red[tid] += red[tid + s];
        __syncthreads();
    }
    float inv = 1.0f / red[0];
    w[(size_t)row * T_ + tid]       = (half_t)(e0 * inv);
    w[(size_t)row * T_ + tid + 256] = (half_t)(e1 * inv);
}

// ---------------------------------------------------------------------------
// GEMM 2 (WMMA): y = w @ x, fused residual: x = x + rs*(y - x); also save to
// the per-layer state buffer.
// One block = 4 waves = one 16-row output slab (it) x all 4 column tiles (nt=wid).
// The 32x64-half xh K-slab (4KB, contiguous) is staged into LDS each K-step
// with the CDNA5 async Global->LDS path (ASYNCcnt), then the per-lane
// K-column gathers for the B operand hit LDS instead of global memory.
// ---------------------------------------------------------------------------
__global__ void attn_apply_gemm(const half_t* __restrict__ wm,
                                const half_t* __restrict__ xh,
                                const float* __restrict__ res_scale, int l,
                                float* x, float* __restrict__ state) {
    __shared__ half_t xs[32 * D_];                // 4KB K-slab
    int tidx = threadIdx.x;                       // 0..127
    int wid  = tidx >> 5;                         // nt = wid (0..3)
    int lane = tidx & 31;
    int it = blockIdx.x & 31;
    int b  = blockIdx.x >> 5;
    int m  = lane & 15, hi = lane >> 4;

    const half_t* arow = wm + ((size_t)b * T_ + it * 16 + m) * T_;
    const half_t* xb   = xh + (size_t)b * T_ * D_;
    int col = wid * 16 + m;
    unsigned lbase = (unsigned)(uintptr_t)xs;     // LDS byte offset of slab

    v8f c = {};
    for (int k0 = 0; k0 < T_; k0 += 32) {
        // --- async stage: 4KB contiguous slab xh[b, k0..k0+31, :] -> LDS ---
        {
            const char* g = (const char*)(xb + (size_t)k0 * D_) + tidx * 16;
            unsigned lo = lbase + tidx * 16;
            asm volatile("global_load_async_to_lds_b128 %0, %1, off"
                         :: "v"(lo), "v"(g) : "memory");
            asm volatile("global_load_async_to_lds_b128 %0, %1, off"
                         :: "v"(lo + 2048u), "v"(g + 2048) : "memory");
        }
        // prefetch next A slab while the async DMA runs
        if (k0 + 32 < T_) __builtin_prefetch(arow + k0 + 32, 0, 0);

        // A operand: contiguous 16B vector loads from wmat row
        v8h alo = *(const v8h*)(arow + k0 + hi * 8);
        v8h ahi = *(const v8h*)(arow + k0 + 16 + hi * 8);
        v16h a;
        #pragma unroll
        for (int e = 0; e < 8; ++e) { a[e] = alo[e]; a[e + 8] = ahi[e]; }

        asm volatile("s_wait_asynccnt 0x0" ::: "memory");
        __syncthreads();

        // B operand: per-lane K-column gather from LDS
        v16h bb;
        #pragma unroll
        for (int e = 0; e < 16; ++e) {
            int kk = ((e < 8) ? e : (e + 8)) + hi * 8;    // 0..31
            bb[e] = xs[kk * D_ + col];
        }
        c = __builtin_amdgcn_wmma_f32_16x16x32_f16(false, a, false, bb,
                                                   (short)0, c, false, false);
        __syncthreads();                          // protect slab before restage
    }
    float rs = res_scale[l];
    #pragma unroll
    for (int r = 0; r < 8; ++r) {
        int M = r + hi * 8;
        size_t idx = ((size_t)b * T_ + it * 16 + M) * D_ + col;
        float xv = x[idx];
        float o  = xv + rs * (c[r] - xv);
        x[idx] = o;
        state[idx] = o;
    }
}

// ---------------------------------------------------------------------------
// Holographic encode: pooled = mean_T(x); hidden = tanh(tanh(pooled@Wc1^T+bc1)@Wc2^T+bc2)
// Single block, 256 threads.
// ---------------------------------------------------------------------------
__global__ void holo_encode(const float* __restrict__ x,
                            const float* __restrict__ Wc1,
                            const float* __restrict__ bc1,
                            const float* __restrict__ Wc2,
                            const float* __restrict__ bc2,
                            float* __restrict__ hidden) {
    __shared__ float pooled[B_ * D_];
    __shared__ float h1[B_ * H_];
    int tid = threadIdx.x;
    if (tid < B_ * D_) {
        int b = tid / D_, d = tid % D_;
        float s = 0.0f;
        for (int t = 0; t < T_; ++t) s += x[((size_t)b * T_ + t) * D_ + d];
        pooled[tid] = s * (1.0f / T_);
    }
    __syncthreads();
    if (tid < B_ * H_) {
        int b = tid / H_, h = tid % H_;
        float a = bc1[h];
        for (int j = 0; j < D_; ++j) a += pooled[b * D_ + j] * Wc1[h * D_ + j];
        h1[tid] = tanhf(a);
    }
    __syncthreads();
    if (tid < B_ * D_) {
        int b = tid / D_, d = tid % D_;
        float a = bc2[d];
        for (int j = 0; j < H_; ++j) a += h1[b * H_ + j] * Wc2[d * H_ + j];
        hidden[tid] = tanhf(a);
    }
}

// ---------------------------------------------------------------------------
// Basin update: agg = mean_B(hidden); g = sigmoid(Wu @ [basin,agg] + bu);
// basin = basin*(1-g) + agg*g.   One block, 64 threads.
// ---------------------------------------------------------------------------
__global__ void basin_update(const float* __restrict__ hidden,
                             const float* __restrict__ Wu,
                             const float* __restrict__ bu,
                             float* basin) {
    int d = threadIdx.x;              // 0..63
    __shared__ float comb[2 * D_];
    float agg = 0.0f;
    for (int b = 0; b < B_; ++b) agg += hidden[b * D_ + d];
    agg *= (1.0f / B_);
    float bv = basin[d];
    comb[d] = bv;
    comb[D_ + d] = agg;
    __syncthreads();
    float acc = bu[d];
    const float* Wr = Wu + (size_t)d * 2 * D_;
    #pragma unroll 8
    for (int j = 0; j < 2 * D_; ++j) acc += Wr[j] * comb[j];
    float g = sigmoidf_(acc);
    float nb = bv * (1.0f - g) + agg * g;
    __syncthreads();
    basin[d] = nb;
}

// ---------------------------------------------------------------------------
// Final: out = x + 0.01 * res_scale_g * (x - basin_seq)
// ---------------------------------------------------------------------------
__global__ void finalize(const float* __restrict__ x,
                         const float* __restrict__ basin_seq,
                         const float* __restrict__ rsg,
                         float* __restrict__ out, int n) {
    int i = blockIdx.x * blockDim.x + threadIdx.x;
    if (i < n) {
        float xv = x[i];
        out[i] = xv + 0.01f * rsg[0] * (xv - basin_seq[i]);
    }
}

// ---------------------------------------------------------------------------
extern "C" void kernel_launch(void* const* d_in, const int* in_sizes, int n_in,
                              void* d_out, int out_size, void* d_ws, size_t ws_size,
                              hipStream_t stream) {
    (void)in_sizes; (void)n_in; (void)out_size; (void)ws_size;
    const float* basin_seq    = (const float*)d_in[0];
    const float* basin_coords = (const float*)d_in[1];
    const float* W_temp       = (const float*)d_in[2];
    const float* b_temp       = (const float*)d_in[3];
    const float* res_scale    = (const float*)d_in[4];
    const float* W_fb         = (const float*)d_in[5];
    const float* b_fb         = (const float*)d_in[6];
    const float* Wc1          = (const float*)d_in[7];
    const float* bc1          = (const float*)d_in[8];
    const float* Wc2          = (const float*)d_in[9];
    const float* bc2          = (const float*)d_in[10];
    const float* Wu           = (const float*)d_in[11];
    const float* bu           = (const float*)d_in[12];
    const float* res_scale_g  = (const float*)d_in[13];
    float* out = (float*)d_out;

    const size_t BTD = (size_t)B_ * T_ * D_;      // 65536
    const size_t BTT = (size_t)B_ * T_ * T_;      // 524288

    // Workspace carve-out (256B aligned slices), ~4.8 MB total
    char* base = (char*)d_ws;
    size_t off = 0;
    auto carve = [&](size_t bytes) -> char* {
        char* p = base + off;
        off = (off + bytes + 255) & ~(size_t)255;
        return p;
    };
    float*  x      = (float*)carve(BTD * sizeof(float));
    float*  states = (float*)carve(L_ * BTD * sizeof(float));
    float*  logits = (float*)carve(BTT * sizeof(float));
    float*  basin  = (float*)carve(D_ * sizeof(float));
    float*  temps  = (float*)carve(L_ * sizeof(float));
    float*  hidden = (float*)carve(B_ * D_ * sizeof(float));
    half_t* sp     = (half_t*)carve(BTD * sizeof(half_t));
    half_t* xh     = (half_t*)carve(BTD * sizeof(half_t));
    half_t* wmat   = (half_t*)carve(BTT * sizeof(half_t));

    // x <- basin_seq ; basin <- basin_coords (device-to-device, capture-safe)
    hipMemcpyAsync(x, basin_seq, BTD * sizeof(float), hipMemcpyDeviceToDevice, stream);
    hipMemcpyAsync(basin, basin_coords, D_ * sizeof(float), hipMemcpyDeviceToDevice, stream);

    const int ROWS = B_ * T_;                     // 1024
    const int G1_TILES = B_ * (T_ / 16) * (T_ / 16);  // 2048
    const int G2_BLOCKS = B_ * (T_ / 16);             // 64 (4 waves each)

    for (int p = 0; p < NPASS_; ++p) {
        temps_kernel<<<1, D_, 0, stream>>>(W_temp, b_temp, basin, temps);
        for (int l = 0; l < L_; ++l) {
            if (p > 0)
                feedback_gate<<<ROWS, D_, 0, stream>>>(x, states + (size_t)l * BTD,
                                                       W_fb, b_fb, l);
            simplex_prep<<<ROWS, D_, 0, stream>>>(x, sp, xh);
            qfi_logits_gemm<<<G1_TILES / 4, 128, 0, stream>>>(sp, temps, l, logits);
            softmax_rows<<<ROWS, 256, 0, stream>>>(logits, wmat);
            attn_apply_gemm<<<G2_BLOCKS, 128, 0, stream>>>(wmat, xh, res_scale, l,
                                                           x, states + (size_t)l * BTD);
        }
        if (p < NPASS_ - 1) {
            holo_encode<<<1, 256, 0, stream>>>(x, Wc1, bc1, Wc2, bc2, hidden);
            basin_update<<<1, D_, 0, stream>>>(hidden, Wu, bu, basin);
        }
    }
    finalize<<<((int)BTD + 255) / 256, 256, 0, stream>>>(x, basin_seq, res_scale_g,
                                                         out, (int)BTD);
}
